// ARDRBFKernel_33483565040048
// MI455X (gfx1250) — matrix operand order repaired and checked by reading, not compiled, not run on hardware
//
#include <hip/hip_runtime.h>
#include <cstdint>
#include <cstddef>

// ---------------------------------------------------------------------------
// ARD-RBF:  out[n,m] = sigma^2 * exp(-0.5 * (sq1[n] + sq2[m] - 2*cross))
//   cross = (x1 * alpha) @ x2^T     (N=M=8192, K=512)
//
// MI455X plan:
//   * bf16 split-precision WMMA GEMM (Ahi*Bhi + Ahi*Blo + Alo*Bhi ~ f32)
//   * K is contiguous for both operands (NT GEMM) -> fragments load directly
//     from global into the CDNA5 WMMA VGPR layout (2x global_load_b128/lane).
//   * D=512 baked in as a template constant so all fragment addresses fold
//     into instruction immediates: one base pointer per operand array,
//     zero address VALU in the k-loop.
//   * 8 waves / block, wave tile 64x64 (4x4 C frags), block tile 128x256:
//     48 WMMA per 32 loads per k-step; bf16 operands (32MB) stay L2-resident.
// ---------------------------------------------------------------------------

typedef __attribute__((ext_vector_type(16))) __bf16 v16bf;
typedef __attribute__((ext_vector_type(8)))  float  v8f;
typedef __attribute__((ext_vector_type(4)))  int    v4i;

struct Frag2 { v4i x, y; };

static __device__ inline unsigned short bf16_rne(float f) {
  unsigned u = __builtin_bit_cast(unsigned, f);
  u += 0x7fffu + ((u >> 16) & 1u);               // round-to-nearest-even
  return (unsigned short)(u >> 16);
}
static __device__ inline float bf16_to_f32(unsigned short h) {
  return __builtin_bit_cast(float, ((unsigned)h) << 16);
}

// ---------------------------------------------------------------------------
// Kernel 1: alpha = softmax(softplus(a_raw)), var = v_raw^2.
// One block, d threads (d = 512, power of two).
// ---------------------------------------------------------------------------
__global__ void ard_alpha_var_kernel(const float* __restrict__ araw,
                                     const float* __restrict__ vraw,
                                     float* __restrict__ alpha,
                                     float* __restrict__ var) {
  __shared__ float red[512];
  const int i = threadIdx.x;
  const int d = blockDim.x;
  float x  = araw[i];
  float sp = fmaxf(x, 0.0f) + log1pf(__expf(-fabsf(x)));   // stable softplus
  red[i] = sp;
  __syncthreads();
  for (int s = d >> 1; s > 0; s >>= 1) {
    if (i < s) red[i] = fmaxf(red[i], red[i + s]);
    __syncthreads();
  }
  float mx = red[0];
  __syncthreads();
  float e = __expf(sp - mx);
  red[i] = e;
  __syncthreads();
  for (int s = d >> 1; s > 0; s >>= 1) {
    if (i < s) red[i] += red[i + s];
    __syncthreads();
  }
  alpha[i] = e / red[0];
  if (i == 0) var[0] = vraw[0] * vraw[0];
}

// ---------------------------------------------------------------------------
// Kernel 2: per-row bf16 hi/lo split + ARD row norm.
// ---------------------------------------------------------------------------
__global__ void ard_prep_kernel(const float* __restrict__ x,
                                const float* __restrict__ alpha,
                                unsigned short* __restrict__ hi,
                                unsigned short* __restrict__ lo,
                                float* __restrict__ sq,
                                int d, int scale_in) {
  __shared__ float red[128];
  const int row = blockIdx.x, tid = threadIdx.x;
  const float* xr = x + (size_t)row * d;
  unsigned short* hr = hi + (size_t)row * d;
  unsigned short* lr = lo + (size_t)row * d;
  float acc = 0.0f;
  for (int i = tid; i < d; i += 128) {
    float xi = xr[i];
    float ai = alpha[i];
    acc = fmaf(ai * xi, xi, acc);
    float v = scale_in ? xi * ai : xi;
    unsigned short h = bf16_rne(v);
    hr[i] = h;
    lr[i] = bf16_rne(v - bf16_to_f32(h));
  }
  red[tid] = acc;
  __syncthreads();
  for (int s = 64; s > 0; s >>= 1) {
    if (tid < s) red[tid] += red[tid + s];
    __syncthreads();
  }
  if (tid == 0) sq[row] = red[0];
}

// ---------------------------------------------------------------------------
// Fragment loaders (wave32):
// A (16x32 16-bit): lane<16 -> M=lane,  K={0..7,16..23}; lane>=16 -> +8
// B (32x16 16-bit): lane<16 -> N=lane,  K={0..7,16..23}; lane>=16 -> +8
// Base pointers already include the per-lane (mrow/hf) component; `off` is a
// compile-time-foldable element offset (mt/nt tile + k step).
// ---------------------------------------------------------------------------
static __device__ inline v16bf load_a_frag(const unsigned short* __restrict__ base,
                                           int off) {
  Frag2 f;
  f.x = *(const v4i*)(base + off);        // K k0+0..7   (+8 for hi half)
  f.y = *(const v4i*)(base + off + 16);   // K k0+16..23 (+8 for hi half)
  return __builtin_bit_cast(v16bf, f);
}
static __device__ inline v16bf load_b_frag(const unsigned short* __restrict__ base,
                                           int off) {
  Frag2 f;
  f.x = *(const v4i*)(base + off);        // K k0+0..7   (+8 for hi half)
  f.y = *(const v4i*)(base + off + 16);   // K k0+16..23 (+8 for hi half)
  return __builtin_bit_cast(v16bf, f);
}

// ---------------------------------------------------------------------------
// Kernel 3: fused split-precision bf16 WMMA GEMM + RBF epilogue.
// Block = 256 threads = 8 waves (2 x 4), block tile = 128 x 256.
// Wave tile = 64 x 64 = 4 x 4 fragments; 48 WMMA / 32 b128-loads per k-step.
// DT > 0: compile-time K (addresses fold to immediates); DT==0: runtime K.
// ---------------------------------------------------------------------------
template <int DT>
__global__ __launch_bounds__(256) void ard_gemm_rbf_kernel(
    const unsigned short* __restrict__ Ahi, const unsigned short* __restrict__ Alo,
    const unsigned short* __restrict__ Bhi, const unsigned short* __restrict__ Blo,
    const float* __restrict__ sq1, const float* __restrict__ sq2,
    const float* __restrict__ varp, float* __restrict__ out,
    int M /*cols = x2 rows*/, int Drt /*K if DT==0*/) {
  const int D = (DT > 0) ? DT : Drt;
  const int lane = threadIdx.x & 31;
  const int wave = threadIdx.x >> 5;
  const int wm = wave >> 2;            // 0..1  (M direction)
  const int wn = wave & 3;             // 0..3  (N direction)
  const int rowBase = blockIdx.y * 128 + wm * 64;
  const int colBase = blockIdx.x * 256 + wn * 64;

  const int mrow = lane & 15;
  const int hf   = lane >> 4;
  // Per-lane base pointers; everything else is a constant element offset.
  const unsigned short* baseAhi = Ahi + (size_t)(rowBase + mrow) * D + hf * 8;
  const unsigned short* baseAlo = Alo + (size_t)(rowBase + mrow) * D + hf * 8;
  const unsigned short* baseBhi = Bhi + (size_t)(colBase + mrow) * D + hf * 8;
  const unsigned short* baseBlo = Blo + (size_t)(colBase + mrow) * D + hf * 8;

  v8f c[4][4] = {};

#pragma unroll 4
  for (int k0 = 0; k0 < D; k0 += 32) {
    v16bf ah[4], al[4];
#pragma unroll
    for (int mt = 0; mt < 4; ++mt) {
      ah[mt] = load_a_frag(baseAhi, mt * 16 * D + k0);
      al[mt] = load_a_frag(baseAlo, mt * 16 * D + k0);
    }
#pragma unroll
    for (int nt = 0; nt < 4; ++nt) {
      const v16bf bh = load_b_frag(baseBhi, nt * 16 * D + k0);
      const v16bf bl = load_b_frag(baseBlo, nt * 16 * D + k0);
#pragma unroll
      for (int mt = 0; mt < 4; ++mt) {
        // cross = Ahi*Bhi + Ahi*Blo + Alo*Bhi  (~f32 accuracy)
        c[mt][nt] = __builtin_amdgcn_wmma_f32_16x16x32_bf16(
            false, ah[mt], false, bh, (short)0, c[mt][nt], false, false);
        c[mt][nt] = __builtin_amdgcn_wmma_f32_16x16x32_bf16(
            false, ah[mt], false, bl, (short)0, c[mt][nt], false, false);
        c[mt][nt] = __builtin_amdgcn_wmma_f32_16x16x32_bf16(
            false, al[mt], false, bh, (short)0, c[mt][nt], false, false);
      }
    }
  }

  // Epilogue: out = var * exp(-0.5 * (sq1[row] + sq2[col] - 2*cross))
  // C/D layout: VGPR r of lane L -> M = r + (L>=16 ? 8 : 0), N = L & 15.
  const float var = varp[0];
#pragma unroll
  for (int mt = 0; mt < 4; ++mt) {
    float s1[8];
#pragma unroll
    for (int r = 0; r < 8; ++r)
      s1[r] = sq1[rowBase + mt * 16 + hf * 8 + r];
#pragma unroll
    for (int nt = 0; nt < 4; ++nt) {
      const int col = colBase + nt * 16 + mrow;
      const float s2 = sq2[col];
#pragma unroll
      for (int r = 0; r < 8; ++r) {
        const int row = rowBase + mt * 16 + hf * 8 + r;
        const float d2 = s1[r] + s2 - 2.0f * c[mt][nt][r];
        out[(size_t)row * M + col] = var * __expf(-0.5f * d2);
      }
    }
  }
}

// ---------------------------------------------------------------------------
// Host launcher
// ---------------------------------------------------------------------------
extern "C" void kernel_launch(void* const* d_in, const int* in_sizes, int n_in,
                              void* d_out, int out_size, void* d_ws, size_t ws_size,
                              hipStream_t stream) {
  const float* x1   = (const float*)d_in[0];
  const float* x2   = (const float*)d_in[1];
  const float* araw = (const float*)d_in[2];
  const float* vraw = (const float*)d_in[3];
  float* out = (float*)d_out;

  const int d = in_sizes[2];          // 512
  const int n = in_sizes[0] / d;      // 8192
  const int m = in_sizes[1] / d;      // 8192

  // Workspace layout (256B-aligned):
  //   alpha[d] | var | sq1[n] | sq2[m] | Ahi,Alo [n*d] bf16 | Bhi,Blo [m*d]
  char* ws = (char*)d_ws;
  size_t off = 0;
  float* alpha = (float*)(ws + off); off += ((size_t)d * 4 + 255) & ~(size_t)255;
  float* var   = (float*)(ws + off); off += 256;
  float* sq1   = (float*)(ws + off); off += ((size_t)n * 4 + 255) & ~(size_t)255;
  float* sq2   = (float*)(ws + off); off += ((size_t)m * 4 + 255) & ~(size_t)255;
  unsigned short* Ahi = (unsigned short*)(ws + off); off += (size_t)n * d * 2;
  unsigned short* Alo = (unsigned short*)(ws + off); off += (size_t)n * d * 2;
  unsigned short* Bhi = (unsigned short*)(ws + off); off += (size_t)m * d * 2;
  unsigned short* Blo = (unsigned short*)(ws + off); off += (size_t)m * d * 2;

  // 1) alpha / variance
  ard_alpha_var_kernel<<<1, d, 0, stream>>>(araw, vraw, alpha, var);
  // 2) bf16 hi/lo split + row norms (alpha folded into A)
  ard_prep_kernel<<<n, 128, 0, stream>>>(x1, alpha, Ahi, Alo, sq1, d, 1);
  ard_prep_kernel<<<m, 128, 0, stream>>>(x2, alpha, Bhi, Blo, sq2, d, 0);
  // 3) fused WMMA GEMM + RBF epilogue: block tile 128 x 256
  dim3 grid(m / 256, n / 128);
  if (d == 512) {
    ard_gemm_rbf_kernel<512><<<grid, 256, 0, stream>>>(
        Ahi, Alo, Bhi, Blo, sq1, sq2, var, out, m, d);
  } else {
    ard_gemm_rbf_kernel<0><<<grid, 256, 0, stream>>>(
        Ahi, Alo, Bhi, Blo, sq1, sq2, var, out, m, d);
  }
}
